// EmbedMatcher_33938831573494
// MI455X (gfx1250) — matrix operand order repaired and checked by reading, not compiled, not run on hardware
//
#include <hip/hip_runtime.h>
#include <hip/hip_bf16.h>

typedef _Float16 h16;
typedef __attribute__((ext_vector_type(16))) _Float16 v16h;
typedef __attribute__((ext_vector_type(8)))  _Float16 v8h;
typedef __attribute__((ext_vector_type(8)))  float    v8f;

#define EE   128
#define DMV  256
#define HIDV 512
#define NB   1024
#define NNB  64

__device__ __forceinline__ v8f zero8() {
  v8f z = {0.f,0.f,0.f,0.f,0.f,0.f,0.f,0.f};
  return z;
}

__device__ __forceinline__ float sigmoidf_(float x) {
  return 1.0f / (1.0f + __expf(-x));
}

// WMMA wrapper: D = A(16x32 f16) x B(32x16 f16) + C(f32)
__device__ __forceinline__ v8f wmma_(v16h a, v16h b, v8f c) {
  return __builtin_amdgcn_wmma_f32_16x16x32_f16(
      /*neg_a=*/false, a, /*neg_b=*/false, b,
      /*c_mod=*/(short)0, c, /*reuse_a=*/false, /*reuse_b=*/false);
}

// A fragment (16x32 f16, wave32 layout) from LDS row-major matrix.
// lane 0-15 : row m, K = kt*32 + {0..15}
// lane 16-31: row m, K = kt*32 + {16..31}
__device__ __forceinline__ v16h load_fragA(const h16* lds, int m, int stride,
                                           int kt, int laneHi) {
  const h16* p = lds + m * stride + kt * 32 + (laneHi ? 16 : 0);
  v8h lo = *(const v8h*)(p);
  v8h hi = *(const v8h*)(p + 8);
  v16h a;
#pragma unroll
  for (int j = 0; j < 8; ++j) { a[j] = lo[j]; a[j + 8] = hi[j]; }
  return a;
}

// -------------------- f32 -> f16 conversion --------------------
__global__ void cvt_kernel(const float* __restrict__ src, h16* __restrict__ dst, int n) {
  int i = blockIdx.x * blockDim.x + threadIdx.x;
  if (i < n) dst[i] = (h16)src[i];
}

// -------------------- neighbor encoder --------------------
// one block per batch row b; 8 waves (256 threads)
__global__ __launch_bounds__(256)
void ne_kernel(const int* __restrict__ conn,       // (B,N,2)
               const int* __restrict__ selfIds,    // (B,2)
               int selfCol,
               const float* __restrict__ emb,      // (NSYM+1,128)
               const h16*  __restrict__ Wh,        // gcn_W f16 (128,256)
               const float* __restrict__ gcn_wb,
               const float* __restrict__ gcn_b,
               const float* __restrict__ attn_W,
               const float* __restrict__ attn_b,
               const float* __restrict__ gate_W,
               const float* __restrict__ gate_wb,
               const float* __restrict__ gate_b,
               float* __restrict__ dst,            // qvec or svec (B,256)
               int colOff, int accumulate) {
  __shared__ __align__(32) h16  Ah[NNB * 2 * EE];     // 64x256 f16 (32 KB)
  __shared__ float projl[NNB * EE];                   // 64x128 f32 (32 KB)
  __shared__ int   relid[NNB], entid[NNB];
  __shared__ float attnw[EE], sc[NNB], wgt[NNB], aggl[EE];
  __shared__ float red[2];

  const int b = blockIdx.x;
  const int tid = threadIdx.x;
  const int lane = tid & 31;
  const int w = tid >> 5;
  const int lane15 = lane & 15;
  const int laneHi = lane >> 4;

  if (tid < NNB) {
    relid[tid] = conn[(b * NNB + tid) * 2 + 0];
    entid[tid] = conn[(b * NNB + tid) * 2 + 1];
  }
  if (tid < EE) attnw[tid] = attn_W[tid];
  __syncthreads();

  // gather + f16 convert: iteration i fetches cat row i (coalesced 2x512B)
#pragma unroll 1
  for (int i = 0; i < NNB; ++i) {
    float v;
    if (tid < EE) v = emb[(size_t)relid[i] * EE + tid];
    else          v = emb[(size_t)entid[i] * EE + (tid - EE)];
    Ah[i * 2 * EE + tid] = (h16)v;
  }
  __syncthreads();

  // GEMM: proj(64x128) = cat(64x256) @ gcn_W^T ; wave w owns e-tile w
  const int e = w * 16 + lane15;
  v16h fb[8];
#pragma unroll
  for (int kt = 0; kt < 8; ++kt)
    fb[kt] = *(const v16h*)(Wh + e * (2 * EE) + kt * 32 + (laneHi ? 16 : 0));

  v8f acc[4];
#pragma unroll
  for (int mi = 0; mi < 4; ++mi) {
    acc[mi] = zero8();
#pragma unroll
    for (int kt = 0; kt < 8; ++kt) {
      v16h fa = load_fragA(Ah, mi * 16 + lane15, 2 * EE, kt, laneHi);
      acc[mi] = wmma_(fa, fb[kt], acc[mi]);
    }
  }
  const float bias = gcn_wb[e] + gcn_b[e];
#pragma unroll
  for (int mi = 0; mi < 4; ++mi) {
#pragma unroll
    for (int r = 0; r < 8; ++r) {
      int m = mi * 16 + r + (laneHi ? 8 : 0);
      float v = acc[mi][r] + bias;
      v = v > 0.f ? v : 0.01f * v;     // leaky_relu
      projl[m * EE + e] = v;
    }
  }
  __syncthreads();

  // attention scores + softmax over n
  if (tid < NNB) {
    float s = attn_b[0];
    for (int j = 0; j < EE; ++j) s += projl[tid * EE + j] * attnw[j];
    sc[tid] = s;
  }
  __syncthreads();
  if (tid == 0) {
    float mx = sc[0];
    for (int n = 1; n < NNB; ++n) mx = fmaxf(mx, sc[n]);
    float sum = 0.f;
    for (int n = 0; n < NNB; ++n) { float ev = __expf(sc[n] - mx); wgt[n] = ev; sum += ev; }
    float inv = 1.f / sum;
    for (int n = 0; n < NNB; ++n) wgt[n] *= inv;
  }
  __syncthreads();
  if (tid < EE) {
    float a = 0.f;
    for (int n = 0; n < NNB; ++n) a += projl[n * EE + tid] * wgt[n];
    aggl[tid] = a;
  }
  __syncthreads();
  if (tid == 0) {
    float g = gate_wb[0] + gate_b[0];
    for (int j = 0; j < EE; ++j) g += aggl[j] * gate_W[j];
    red[0] = sigmoidf_(g);
  }
  __syncthreads();
  if (tid < EE) {
    float g = red[0];
    int sid = selfIds[b * 2 + selfCol];
    float res = 0.5f * (g * aggl[tid] + (1.f - g) * emb[(size_t)sid * EE + tid]);
    float* p = dst + (size_t)b * DMV + colOff + tid;
    if (accumulate) *p += res; else *p = res;
  }
}

// -------------------- support encoder --------------------
// block = 16 batch rows, 8 waves
__global__ __launch_bounds__(256)
void se_kernel(const float* __restrict__ x,     // svec (B,256)
               const h16*  __restrict__ W1h,    // (512,256)
               const float* __restrict__ b1,
               const h16*  __restrict__ W2h,    // (256,512)
               const float* __restrict__ b2,
               const float* __restrict__ ln_g,
               const float* __restrict__ ln_b,
               float* __restrict__ sg) {        // (B,256)
  __shared__ __align__(32) h16 Ah[16 * DMV];        // 8 KB
  __shared__ float xf[16 * DMV];                    // 16 KB
  __shared__ __align__(32) h16 Hh[16 * 2 * DMV];    // 16 KB
  __shared__ float outf[16 * DMV];                  // 16 KB
  __shared__ float stat[32];

  const int r0 = blockIdx.x * 16;
  const int tid = threadIdx.x;
  const int lane = tid & 31, w = tid >> 5;
  const int lane15 = lane & 15, laneHi = lane >> 4;

  for (int idx = tid; idx < 16 * DMV; idx += 256) {
    float v = x[(size_t)(r0 + idx / DMV) * DMV + (idx % DMV)];
    xf[idx] = v;
    Ah[idx] = (h16)v;
  }
  __syncthreads();

  // GEMM1: h(16x512) = relu(x @ W1^T + b1)
#pragma unroll
  for (int t = 0; t < 4; ++t) {
    int n = (w + 8 * t) * 16 + lane15;
    v8f acc = zero8();
#pragma unroll
    for (int kt = 0; kt < 8; ++kt) {
      v16h fbv = *(const v16h*)(W1h + n * DMV + kt * 32 + (laneHi ? 16 : 0));
      v16h fa  = load_fragA(Ah, lane15, DMV, kt, laneHi);
      acc = wmma_(fa, fbv, acc);
    }
    float bias = b1[n];
#pragma unroll
    for (int r = 0; r < 8; ++r) {
      int m = r + (laneHi ? 8 : 0);
      Hh[m * (2 * DMV) + n] = (h16)fmaxf(acc[r] + bias, 0.f);
    }
  }
  __syncthreads();

  // GEMM2: out(16x256) = h @ W2^T + b2 + x
#pragma unroll
  for (int t = 0; t < 2; ++t) {
    int n = (w + 8 * t) * 16 + lane15;
    v8f acc = zero8();
#pragma unroll
    for (int kt = 0; kt < 16; ++kt) {
      v16h fbv = *(const v16h*)(W2h + n * (2 * DMV) + kt * 32 + (laneHi ? 16 : 0));
      v16h fa  = load_fragA(Hh, lane15, 2 * DMV, kt, laneHi);
      acc = wmma_(fa, fbv, acc);
    }
    float bias = b2[n];
#pragma unroll
    for (int r = 0; r < 8; ++r) {
      int m = r + (laneHi ? 8 : 0);
      outf[m * DMV + n] = acc[r] + bias + xf[m * DMV + n];
    }
  }
  __syncthreads();

  if (tid < 16) {
    float mu = 0.f;
    for (int j = 0; j < DMV; ++j) mu += outf[tid * DMV + j];
    mu *= (1.f / DMV);
    float var = 0.f;
    for (int j = 0; j < DMV; ++j) { float d = outf[tid * DMV + j] - mu; var += d * d; }
    var *= (1.f / DMV);
    stat[tid * 2] = mu;
    stat[tid * 2 + 1] = rsqrtf(var + 1e-6f);
  }
  __syncthreads();
  for (int idx = tid; idx < 16 * DMV; idx += 256) {
    int m = idx / DMV, j = idx % DMV;
    float v = (outf[idx] - stat[m * 2]) * stat[m * 2 + 1];
    sg[(size_t)(r0 + m) * DMV + j] = ln_g[j] * v + ln_b[j];
  }
}

// -------------------- step-invariant LSTM input gates --------------------
__global__ __launch_bounds__(256)
void gih_kernel(const float* __restrict__ qv,   // (B,256)
                const h16*  __restrict__ Wih,   // (2048,256)
                const float* __restrict__ bih,
                const float* __restrict__ bhh,
                float* __restrict__ gih) {      // (B,2048)
  __shared__ __align__(32) h16 Ah[16 * DMV];
  const int r0 = blockIdx.x * 16;
  const int tid = threadIdx.x;
  const int lane = tid & 31, w = tid >> 5;
  const int lane15 = lane & 15, laneHi = lane >> 4;

  for (int idx = tid; idx < 16 * DMV; idx += 256)
    Ah[idx] = (h16)qv[(size_t)(r0 + idx / DMV) * DMV + (idx % DMV)];
  __syncthreads();

#pragma unroll 1
  for (int t = 0; t < 16; ++t) {
    int n = (w + 8 * t) * 16 + lane15;
    v8f acc = zero8();
#pragma unroll
    for (int kt = 0; kt < 8; ++kt) {
      v16h fbv = *(const v16h*)(Wih + n * DMV + kt * 32 + (laneHi ? 16 : 0));
      v16h fa  = load_fragA(Ah, lane15, DMV, kt, laneHi);
      acc = wmma_(fa, fbv, acc);
    }
    float bias = bih[n] + bhh[n];
#pragma unroll
    for (int r = 0; r < 8; ++r) {
      int m = r + (laneHi ? 8 : 0);
      gih[(size_t)(r0 + m) * 2048 + n] = acc[r] + bias;
    }
  }
}

// -------------------- LSTM recurrence + final dot --------------------
__global__ __launch_bounds__(256)
void lstm_kernel(const float* __restrict__ gih,  // (B,2048)
                 const h16*  __restrict__ Whh,   // (2048,512)
                 const float* __restrict__ qv,   // (B,256)
                 const float* __restrict__ sg,   // (B,256)
                 float* __restrict__ out) {      // (B,)
  __shared__ __align__(32) h16 hrh[16 * HIDV];   // 16 KB
  __shared__ float qvf[16 * DMV];                // 16 KB
  __shared__ float hf[16 * DMV];                 // 16 KB

  const int r0 = blockIdx.x * 16;
  const int tid = threadIdx.x;
  const int lane = tid & 31, w = tid >> 5;
  const int lane15 = lane & 15, laneHi = lane >> 4;
  const int mOff = laneHi ? 8 : 0;

  for (int idx = tid; idx < 16 * DMV; idx += 256)
    qvf[idx] = qv[(size_t)(r0 + idx / DMV) * DMV + (idx % DMV)];
  for (int idx = tid; idx < 16 * HIDV; idx += 256)
    hrh[idx] = (h16)0.f;                         // hr starts at zero
  __syncthreads();

  float c[4][8];
#pragma unroll
  for (int jt = 0; jt < 4; ++jt)
#pragma unroll
    for (int r = 0; r < 8; ++r) c[jt][r] = 0.f;

#pragma unroll 1
  for (int s = 0; s < 4; ++s) {
    float hval[2][8];
#pragma unroll
    for (int jt = 0; jt < 4; ++jt) {
      v8f a4[4];
#pragma unroll
      for (int gidx = 0; gidx < 4; ++gidx) {     // 0:i 1:f 2:g 3:o
        int n = (gidx * 32 + w + 8 * jt) * 16 + lane15;
        v8f acc;
#pragma unroll
        for (int r = 0; r < 8; ++r)
          acc[r] = gih[(size_t)(r0 + r + mOff) * 2048 + n];
#pragma unroll
        for (int kt = 0; kt < 16; ++kt) {
          v16h fbv = *(const v16h*)(Whh + n * HIDV + kt * 32 + (laneHi ? 16 : 0));
          v16h fa  = load_fragA(hrh, lane15, HIDV, kt, laneHi);
          acc = wmma_(fa, fbv, acc);
        }
        a4[gidx] = acc;
      }
#pragma unroll
      for (int r = 0; r < 8; ++r) {
        float cc = c[jt][r];
        cc = sigmoidf_(a4[1][r]) * cc + sigmoidf_(a4[0][r]) * tanhf(a4[2][r]);
        c[jt][r] = cc;
        float hc = sigmoidf_(a4[3][r]) * tanhf(cc);
        if (jt < 2) {
          int j = (w + 8 * jt) * 16 + lane15;
          hval[jt][r] = qvf[(r + mOff) * DMV + j] + hc;
        }
      }
    }
    __syncthreads();   // all hrh reads of this step are done
#pragma unroll
    for (int jt = 0; jt < 2; ++jt) {
      int j = (w + 8 * jt) * 16 + lane15;
#pragma unroll
      for (int r = 0; r < 8; ++r) {
        int m = r + mOff;
        hf[m * DMV + j] = hval[jt][r];
        hrh[m * HIDV + j] = (h16)hval[jt][r];
      }
    }
    for (int idx = tid; idx < 16 * DMV; idx += 256) {   // hr right half = support_g
      int m = idx / DMV, j = idx % DMV;
      hrh[m * HIDV + DMV + j] = (h16)sg[(size_t)(r0 + m) * DMV + j];
    }
    __syncthreads();
  }

  if (tid < 16) {
    float d = 0.f;
    for (int j = 0; j < DMV; ++j)
      d += hf[tid * DMV + j] * sg[(size_t)(r0 + tid) * DMV + j];
    out[r0 + tid] = d;
  }
}

// -------------------- host launcher --------------------
extern "C" void kernel_launch(void* const* d_in, const int* in_sizes, int n_in,
                              void* d_out, int out_size, void* d_ws, size_t ws_size,
                              hipStream_t stream) {
  const int* query   = (const int*)d_in[0];
  const int* support = (const int*)d_in[1];
  const int* conns[8];
  for (int i = 0; i < 8; ++i) conns[i] = (const int*)d_in[2 + i];
  const float* emb    = (const float*)d_in[14];
  const float* gcn_W  = (const float*)d_in[15];
  const float* gcn_wb = (const float*)d_in[16];
  const float* gcn_b  = (const float*)d_in[17];
  const float* attn_W = (const float*)d_in[18];
  const float* attn_b = (const float*)d_in[19];
  const float* gate_W = (const float*)d_in[20];
  const float* gate_wb= (const float*)d_in[21];
  const float* gate_b = (const float*)d_in[22];
  const float* se_W1  = (const float*)d_in[23];
  const float* se_b1  = (const float*)d_in[24];
  const float* se_W2  = (const float*)d_in[25];
  const float* se_b2  = (const float*)d_in[26];
  const float* ln_g   = (const float*)d_in[27];
  const float* ln_b   = (const float*)d_in[28];
  const float* W_ih   = (const float*)d_in[29];
  const float* W_hh   = (const float*)d_in[30];
  const float* b_ih   = (const float*)d_in[31];
  const float* b_hh   = (const float*)d_in[32];

  char* ws = (char*)d_ws;
  size_t o = 0;
  h16* gcnWh = (h16*)(ws + o); o += (size_t)128 * 256 * 2;
  h16* W1h   = (h16*)(ws + o); o += (size_t)512 * 256 * 2;
  h16* W2h   = (h16*)(ws + o); o += (size_t)256 * 512 * 2;
  h16* Wihh  = (h16*)(ws + o); o += (size_t)2048 * 256 * 2;
  h16* Whhh  = (h16*)(ws + o); o += (size_t)2048 * 512 * 2;
  float* qvec = (float*)(ws + o); o += (size_t)NB * DMV * 4;
  float* svec = (float*)(ws + o); o += (size_t)NB * DMV * 4;
  float* sgv  = (float*)(ws + o); o += (size_t)NB * DMV * 4;
  float* gih  = (float*)(ws + o); o += (size_t)NB * 2048 * 4;

  // weight conversions (layouts are already n-major, plain copy)
  auto cvt = [&](const float* s, h16* d, int n) {
    cvt_kernel<<<(n + 255) / 256, 256, 0, stream>>>(s, d, n);
  };
  cvt(gcn_W, gcnWh, 128 * 256);
  cvt(se_W1, W1h,   512 * 256);
  cvt(se_W2, W2h,   256 * 512);
  cvt(W_ih,  Wihh,  2048 * 256);
  cvt(W_hh,  Whhh,  2048 * 512);

  // 8 neighbor-encoder passes: {conn idx, selfCol, dst, colOff, accumulate}
  struct { int ci; const int* self; int scol; float* dst; int coff; int acc; } ne8[8] = {
    {0, query,   0, qvec,   0, 0}, {1, query,   0, qvec,   0, 1},
    {2, query,   1, qvec, 128, 0}, {3, query,   1, qvec, 128, 1},
    {4, support, 0, svec,   0, 0}, {5, support, 0, svec,   0, 1},
    {6, support, 1, svec, 128, 0}, {7, support, 1, svec, 128, 1},
  };
  for (int i = 0; i < 8; ++i) {
    ne_kernel<<<NB, 256, 0, stream>>>(
        conns[ne8[i].ci], ne8[i].self, ne8[i].scol, emb, gcnWh,
        gcn_wb, gcn_b, attn_W, attn_b, gate_W, gate_wb, gate_b,
        ne8[i].dst, ne8[i].coff, ne8[i].acc);
  }

  se_kernel<<<NB / 16, 256, 0, stream>>>(svec, W1h, se_b1, W2h, se_b2,
                                         ln_g, ln_b, sgv);
  gih_kernel<<<NB / 16, 256, 0, stream>>>(qvec, Wihh, b_ih, b_hh, gih);
  lstm_kernel<<<NB / 16, 256, 0, stream>>>(gih, Whhh, qvec, sgv,
                                           (float*)d_out);
  (void)in_sizes; (void)n_in; (void)out_size; (void)ws_size;
}
